// Perf1Transformer_8675833938627
// MI455X (gfx1250) — compile-verified, matrix operand order checked by reading, not run on hardware
//
#include <hip/hip_runtime.h>
#include <hip/hip_bf16.h>

#define BB 2
#define SS 1024
#define VV 50257
#define DD 512
#define HH 8
#define DHH 64
#define LL 4
#define DMM 2048

typedef __bf16 bf16_t;
typedef bf16_t bfx16 __attribute__((ext_vector_type(16)));
typedef bf16_t bfx8  __attribute__((ext_vector_type(8)));
typedef float  fx8   __attribute__((ext_vector_type(8)));

enum { EP_F32 = 0, EP_BF = 1, EP_SCORES = 2, EP_GELU_BF = 3, EP_RES_F32 = 4, EP_RES_F32_BF = 5 };

__device__ __forceinline__ bf16_t f2bf(float f) { return (bf16_t)f; }

// Async copy of 16 bytes global -> LDS (per active lane). Tracked by ASYNCcnt.
__device__ __forceinline__ void async_ld16(unsigned lds_off, const bf16_t* g)
{
  asm volatile("global_load_async_to_lds_b128 %0, %1, off"
               :: "v"(lds_off), "v"((unsigned long long)(uintptr_t)g)
               : "memory");
}
__device__ __forceinline__ void wait_async0()
{
  asm volatile("s_wait_asynccnt 0" ::: "memory");
}

__device__ __forceinline__ bfx16 load_frag_g(const bf16_t* p)
{
  // ISA 16-bit A layout: lane half 0 -> K {0..7,16..23}; half 1 -> K {8..15,24..31}
  bfx8 a0 = *(const bfx8*)(p);
  bfx8 a1 = *(const bfx8*)(p + 16);
  bfx16 r;
#pragma unroll
  for (int i = 0; i < 8; ++i) { r[i] = a0[i]; r[8 + i] = a1[i]; }
  return r;
}

// ---------------------------------------------------------------------------
// LDS-staged batched bf16 GEMM:  C[z][m][n] = sum_k A[z][m][k] * W[z][n][k]
//   A: bf16 row-major [M,K]; W: bf16 [N,K] row-major (pre-transposed B).
//   WG = 128 threads (4 waves). Macro tile 128x64, BK=32.
//   B tile double-buffered in LDS via global_load_async_to_lds_b128;
//   each wave computes 32x64 via 8x v_wmma_f32_16x16x32_bf16 per K-step.
// ---------------------------------------------------------------------------
template <int EP>
__global__ __launch_bounds__(128) void k_gemm_lds(
    const bf16_t* __restrict__ A, long lda, long sAb, long sAh,
    const bf16_t* __restrict__ W, long ldw, long sWb, long sWh,
    int M, int N, int K, int Hdim,
    float* __restrict__ Cf, bf16_t* __restrict__ Cb,
    long ldcm, long ldcn, long sCb, long sCh,
    const float* __restrict__ R, float scale)
{
  constexpr int BK   = 32;
  constexpr int BSTR = 40;                    // padded LDS row stride (elems) = 80B
  __shared__ bf16_t sB[2][64 * BSTR];         // 10 KB

  const int z  = blockIdx.z;
  const int zb = (Hdim > 1) ? (z / Hdim) : z;
  const int zh = (Hdim > 1) ? (z % Hdim) : 0;
  A += (long)zb * sAb + (long)zh * sAh;
  W += (long)zb * sWb + (long)zh * sWh;
  const long coff = (long)zb * sCb + (long)zh * sCh;

  const int tid  = threadIdx.x;
  const int wave = tid >> 5;
  const int lane = tid & 31;
  const int half = lane >> 4;
  const int l16  = lane & 15;

  const int tn0 = blockIdx.x * 64;
  const int tm0 = blockIdx.y * 128 + wave * 32;

  // --- B-tile async loader: 64 rows x 32 k = 4KB/step, 2x b128 per thread ---
  const unsigned sb_base = (unsigned)(uintptr_t)(&sB[0][0]);
  const bf16_t* gB[2];
  unsigned      dB[2];
#pragma unroll
  for (int j = 0; j < 2; ++j) {
    const int li  = tid * 2 + j;              // 0..255
    const int row = li >> 2;                  // 0..63
    const int ch  = li & 3;                   // 16B chunk within the 64B row
    int col = tn0 + row;
    if (col >= N) col = N - 1;                // clamp; stores are guarded
    gB[j] = W + (long)col * ldw + ch * 8;
    dB[j] = sb_base + (unsigned)(row * BSTR + ch * 8) * 2u;
  }
  auto issueB = [&](int buf, int k0) {
#pragma unroll
    for (int j = 0; j < 2; ++j)
      async_ld16(dB[j] + (unsigned)buf * (64u * BSTR * 2u), gB[j] + k0);
  };

  // --- A fragment pointers (direct global; rows private to this wave) ---
  const bf16_t* aptr0 = A + (long)(tm0 + l16) * lda + half * 8;
  const bf16_t* aptr1 = A + (long)(tm0 + 16 + l16) * lda + half * 8;

  fx8 acc[2][4];
#pragma unroll
  for (int mt = 0; mt < 2; ++mt)
#pragma unroll
    for (int nt = 0; nt < 4; ++nt) acc[mt][nt] = fx8{};

  // One K-step: clause of 8 ds_load_b128 into private frags, then 8 WMMAs.
  auto compute = [&](int ks) {
    const int k0 = ks * BK;
    const bfx16 a0 = load_frag_g(aptr0 + k0);
    const bfx16 a1 = load_frag_g(aptr1 + k0);
    const bf16_t* sb = &sB[ks & 1][0];
    bfx16 bfrag[4];
#pragma unroll
    for (int nt = 0; nt < 4; ++nt) {
      const bf16_t* bp = sb + (nt * 16 + l16) * BSTR + half * 8;
      bfx8 b0 = *(const bfx8*)(bp);
      bfx8 b1 = *(const bfx8*)(bp + 16);
#pragma unroll
      for (int i = 0; i < 8; ++i) { bfrag[nt][i] = b0[i]; bfrag[nt][8 + i] = b1[i]; }
    }
#pragma unroll
    for (int nt = 0; nt < 4; ++nt) {
      acc[0][nt] = __builtin_amdgcn_wmma_f32_16x16x32_bf16(
          false, a0, false, bfrag[nt], (short)0, acc[0][nt], false, false);
      acc[1][nt] = __builtin_amdgcn_wmma_f32_16x16x32_bf16(
          false, a1, false, bfrag[nt], (short)0, acc[1][nt], false, false);
    }
  };

  issueB(0, 0);
  const int nk = K / BK;
  for (int ks = 0; ks < nk - 1; ++ks) {       // branch-free steady state
    wait_async0();                            // our async writes have landed
    __syncthreads();                          // all writes visible, reads retired
    issueB((ks + 1) & 1, (ks + 1) * BK);      // prefetch next tile
    compute(ks);
  }
  wait_async0();                              // peeled last K-step
  __syncthreads();
  compute(nk - 1);

  // --- epilogue: acc elem i -> row tm0 + mt*16 + half*8 + i, col tile l16 ---
#pragma unroll
  for (int mt = 0; mt < 2; ++mt) {
#pragma unroll
    for (int nt = 0; nt < 4; ++nt) {
      const int n = tn0 + nt * 16 + l16;
      if (n >= N) continue;
#pragma unroll
      for (int i = 0; i < 8; ++i) {
        const int m = tm0 + mt * 16 + half * 8 + i;
        if (m >= M) continue;
        float v = acc[mt][nt][i];
        const long idx = coff + (long)m * ldcm + (long)n * ldcn;
        if (EP == EP_SCORES) {
          v *= scale;
          if (n > m) v = -3.0e38f;            // causal mask
          Cf[idx] = v;
        } else if (EP == EP_F32) {
          Cf[idx] = v;
        } else if (EP == EP_BF) {
          Cb[idx] = f2bf(v);
        } else if (EP == EP_GELU_BF) {
          float g = 0.5f * v * (1.0f + erff(v * 0.70710678118654752f));
          Cb[idx] = f2bf(g);
        } else if (EP == EP_RES_F32) {
          Cf[idx] = R[idx] + v;
        } else if (EP == EP_RES_F32_BF) {
          float s = R[idx] + v;
          Cf[idx] = s;
          Cb[idx] = f2bf(s);
        }
      }
    }
  }
}

// ---------------------------------------------------------------------------
// Embedding gather + positional add; writes fp32 h and bf16 copy.
// ---------------------------------------------------------------------------
__global__ void k_embed(const int* __restrict__ x, const float* __restrict__ em,
                        const float* __restrict__ pos,
                        float* __restrict__ h, bf16_t* __restrict__ hb)
{
  const long idx = (long)blockIdx.x * 256 + threadIdx.x;  // BB*SS*DD total
  const int row = (int)(idx >> 9);
  const int d   = (int)(idx & (DD - 1));
  const int s   = row & (SS - 1);
  const float v = em[(long)x[row] * DD + d] + pos[(long)s * DD + d];
  h[idx]  = v;
  hb[idx] = f2bf(v);
}

// ---------------------------------------------------------------------------
// Row softmax over S=1024 fp32 scores -> bf16 probabilities. 256 threads/row.
// ---------------------------------------------------------------------------
__global__ void k_softmax(const float* __restrict__ sc, bf16_t* __restrict__ pr)
{
  const long row = blockIdx.x;
  const float* p = sc + row * (long)SS;
  __shared__ float red[256];
  const int t = threadIdx.x;

  float v0 = p[t], v1 = p[t + 256], v2 = p[t + 512], v3 = p[t + 768];
  float mx = fmaxf(fmaxf(v0, v1), fmaxf(v2, v3));
  red[t] = mx; __syncthreads();
  for (int st = 128; st > 0; st >>= 1) {
    if (t < st) red[t] = fmaxf(red[t], red[t + st]);
    __syncthreads();
  }
  mx = red[0];
  __syncthreads();

  const float L2E = 1.4426950408889634f;
  float e0 = exp2f((v0 - mx) * L2E);
  float e1 = exp2f((v1 - mx) * L2E);
  float e2 = exp2f((v2 - mx) * L2E);
  float e3 = exp2f((v3 - mx) * L2E);
  red[t] = e0 + e1 + e2 + e3; __syncthreads();
  for (int st = 128; st > 0; st >>= 1) {
    if (t < st) red[t] += red[t + st];
    __syncthreads();
  }
  const float inv = 1.0f / red[0];

  bf16_t* o = pr + row * (long)SS;
  o[t]       = f2bf(e0 * inv);
  o[t + 256] = f2bf(e1 * inv);
  o[t + 512] = f2bf(e2 * inv);
  o[t + 768] = f2bf(e3 * inv);
}

// ---------------------------------------------------------------------------
// LayerNorm over D=512 per row; writes fp32 and bf16. 256 threads/row.
// ---------------------------------------------------------------------------
__global__ void k_layernorm(const float* __restrict__ in, const float* __restrict__ g,
                            const float* __restrict__ b,
                            float* __restrict__ out, bf16_t* __restrict__ outb)
{
  const int row = blockIdx.x;
  const float* p = in + (long)row * DD;
  __shared__ float red[256];
  const int t = threadIdx.x;

  float a0 = p[t], a1 = p[t + 256];
  red[t] = a0 + a1; __syncthreads();
  for (int st = 128; st > 0; st >>= 1) {
    if (t < st) red[t] += red[t + st];
    __syncthreads();
  }
  const float mu = red[0] * (1.0f / DD);
  __syncthreads();

  const float d0 = a0 - mu, d1 = a1 - mu;
  red[t] = d0 * d0 + d1 * d1; __syncthreads();
  for (int st = 128; st > 0; st >>= 1) {
    if (t < st) red[t] += red[t + st];
    __syncthreads();
  }
  const float rs = rsqrtf(red[0] * (1.0f / DD) + 1e-5f);

  const float o0 = d0 * rs * g[t] + b[t];
  const float o1 = d1 * rs * g[t + 256] + b[t + 256];
  out[(long)row * DD + t]        = o0;
  out[(long)row * DD + t + 256]  = o1;
  outb[(long)row * DD + t]       = f2bf(o0);
  outb[(long)row * DD + t + 256] = f2bf(o1);
}

// ---------------------------------------------------------------------------
// Weight conversion kernels.
// ---------------------------------------------------------------------------
__global__ void k_cvt(const float* __restrict__ in, bf16_t* __restrict__ out, long n)
{
  const long i = (long)blockIdx.x * 256 + threadIdx.x;
  if (i < n) out[i] = f2bf(in[i]);
}

// in [rows, cols] fp32 -> out [cols, rows] bf16
__global__ void k_transpose_cvt(const float* __restrict__ in, bf16_t* __restrict__ out,
                                int rows, int cols)
{
  const long i = (long)blockIdx.x * 256 + threadIdx.x;
  if (i < (long)rows * cols) {
    const int r = (int)(i / cols), c = (int)(i % cols);
    out[(long)c * rows + r] = f2bf(in[i]);
  }
}

// ===========================================================================
extern "C" void kernel_launch(void* const* d_in, const int* in_sizes, int n_in,
                              void* d_out, int out_size, void* d_ws, size_t ws_size,
                              hipStream_t stream)
{
  const int*   x    = (const int*)  d_in[0];
  const float* em   = (const float*)d_in[1];
  const float* pos  = (const float*)d_in[2];
  const float* Qw   = (const float*)d_in[3];   // [L, D, H*DH]
  const float* Kw   = (const float*)d_in[4];   // [L, D, H*DH]
  const float* Vw   = (const float*)d_in[5];   // [L, H, D, DH]
  const float* Ow   = (const float*)d_in[6];   // [L, H*DH, D]
  const float* Wi   = (const float*)d_in[7];   // [L, D, DM]
  const float* Wo   = (const float*)d_in[8];   // [L, DM, D]
  const float* gam  = (const float*)d_in[9];   // [L, D]
  const float* bet  = (const float*)d_in[10];  // [L, D]
  float* out = (float*)d_out;

  // ---- workspace carve ----
  size_t off = 0;
  auto carve = [&](size_t elems, size_t esz) -> void* {
    void* p = (char*)d_ws + off;
    off += (elems * esz + 255) & ~(size_t)255;
    return p;
  };
  const size_t MT = (size_t)BB * SS;                 // 2048 rows
  bf16_t* embB = (bf16_t*)carve((size_t)VV * DD, 2); // [V, D] = [N,K] for unembed
  bf16_t* qWt  = (bf16_t*)carve((size_t)LL * DD * DD, 2);       // [L][512][512]
  bf16_t* kWt  = (bf16_t*)carve((size_t)LL * DD * DD, 2);
  bf16_t* vWt  = (bf16_t*)carve((size_t)LL * DD * DD, 2);       // [L][H*DH][D]
  bf16_t* oWt  = (bf16_t*)carve((size_t)LL * DD * DD, 2);
  bf16_t* wiT  = (bf16_t*)carve((size_t)LL * DMM * DD, 2);      // [L][DM][D]
  bf16_t* woT  = (bf16_t*)carve((size_t)LL * DD * DMM, 2);      // [L][D][DM]
  float*  h    = (float*) carve(MT * DD, 4);
  bf16_t* hb   = (bf16_t*)carve(MT * DD, 2);
  float*  tmp  = (float*) carve(MT * DD, 4);
  bf16_t* qb   = (bf16_t*)carve(MT * DD, 2);         // [B,S,H*DH]
  bf16_t* kb   = (bf16_t*)carve(MT * DD, 2);
  bf16_t* vt   = (bf16_t*)carve(MT * DD, 2);         // [B,H,DH,S]
  bf16_t* zb   = (bf16_t*)carve(MT * DD, 2);         // [B,S,H*DH]
  float*  scr  = (float*) carve((size_t)BB * HH * SS * SS, 4);
  bf16_t* prb  = (bf16_t*)carve((size_t)BB * HH * SS * SS, 2);
  bf16_t* act  = (bf16_t*)carve(MT * DMM, 2);        // [2048, 2048]
  (void)ws_size; (void)n_in; (void)in_sizes; (void)out_size;

  // ---- weight conversion ----
  {
    long n = (long)VV * DD;
    k_cvt<<<dim3((unsigned)((n + 255) / 256)), 256, 0, stream>>>(em, embB, n);
  }
  for (int l = 0; l < LL; ++l) {
    const long sq = (long)l * DD * DD;
    unsigned g512 = (unsigned)(((long)DD * DD + 255) / 256);
    k_transpose_cvt<<<g512, 256, 0, stream>>>(Qw + sq, qWt + sq, DD, DD);
    k_transpose_cvt<<<g512, 256, 0, stream>>>(Kw + sq, kWt + sq, DD, DD);
    k_transpose_cvt<<<g512, 256, 0, stream>>>(Ow + sq, oWt + sq, DD, DD);
    for (int hh = 0; hh < HH; ++hh) {
      // Vw[l][hh]: [D, DH] -> vWt row block [hh*DH .. hh*DH+DH) x [D]
      k_transpose_cvt<<<(unsigned)(((long)DD * DHH + 255) / 256), 256, 0, stream>>>(
          Vw + ((long)(l * HH + hh) * DD * DHH),
          vWt + sq + (long)hh * DHH * DD, DD, DHH);
    }
    unsigned g2048 = (unsigned)(((long)DD * DMM + 255) / 256);
    k_transpose_cvt<<<g2048, 256, 0, stream>>>(Wi + (long)l * DD * DMM,
                                               wiT + (long)l * DMM * DD, DD, DMM);
    k_transpose_cvt<<<g2048, 256, 0, stream>>>(Wo + (long)l * DMM * DD,
                                               woT + (long)l * DD * DMM, DMM, DD);
  }

  // ---- embedding ----
  k_embed<<<(unsigned)((MT * DD) / 256), 256, 0, stream>>>(x, em, pos, h, hb);

  const float iscale = 0.125f;  // 1/sqrt(DH)

  for (int l = 0; l < LL; ++l) {
    const bf16_t* qw = qWt + (long)l * DD * DD;
    const bf16_t* kw = kWt + (long)l * DD * DD;
    const bf16_t* vw = vWt + (long)l * DD * DD;
    const bf16_t* ow = oWt + (long)l * DD * DD;
    const bf16_t* wi = wiT + (long)l * DMM * DD;
    const bf16_t* wo = woT + (long)l * DD * DMM;

    // q = h @ Q  -> bf16 [B*S, 512]
    k_gemm_lds<EP_BF><<<dim3(DD / 64, MT / 128, 1), 128, 0, stream>>>(
        hb, DD, 0, 0, qw, DD, 0, 0, (int)MT, DD, DD, 1,
        nullptr, qb, DD, 1, 0, 0, nullptr, 0.0f);
    // k = h @ K
    k_gemm_lds<EP_BF><<<dim3(DD / 64, MT / 128, 1), 128, 0, stream>>>(
        hb, DD, 0, 0, kw, DD, 0, 0, (int)MT, DD, DD, 1,
        nullptr, kb, DD, 1, 0, 0, nullptr, 0.0f);
    // v = h @ Vw, stored transposed [B,H,DH,S] (n*S + m per batch)
    k_gemm_lds<EP_BF><<<dim3(DD / 64, SS / 128, BB), 128, 0, stream>>>(
        hb, DD, (long)SS * DD, 0, vw, DD, 0, 0, SS, DD, DD, 1,
        nullptr, vt, 1, SS, (long)HH * DHH * SS, 0, nullptr, 0.0f);

    // scores[b,h] = q k^T * scale + causal mask  (fp32 [B,H,S,S])
    k_gemm_lds<EP_SCORES><<<dim3(SS / 64, SS / 128, BB * HH), 128, 0, stream>>>(
        qb, DD, (long)SS * DD, DHH, kb, DD, (long)SS * DD, DHH, SS, SS, DHH, HH,
        scr, nullptr, SS, 1, (long)HH * SS * SS, (long)SS * SS, nullptr, iscale);

    // softmax rows -> bf16 probs
    k_softmax<<<(unsigned)(BB * HH * SS), 256, 0, stream>>>(scr, prb);

    // z[b,h] = probs @ v  -> [B,S,H*DH] bf16
    k_gemm_lds<EP_BF><<<dim3(DHH / 64, SS / 128, BB * HH), 128, 0, stream>>>(
        prb, SS, (long)HH * SS * SS, (long)SS * SS,
        vt, SS, (long)HH * DHH * SS, (long)DHH * SS, SS, DHH, SS, HH,
        nullptr, zb, DD, 1, (long)SS * DD, DHH, nullptr, 0.0f);

    // tmp = h + z @ O   (fp32, pre-LN)
    k_gemm_lds<EP_RES_F32><<<dim3(DD / 64, MT / 128, 1), 128, 0, stream>>>(
        zb, DD, 0, 0, ow, DD, 0, 0, (int)MT, DD, DD, 1,
        tmp, nullptr, DD, 1, 0, 0, h, 0.0f);

    // h = LN(tmp) -> fp32 h + bf16 hb
    k_layernorm<<<(unsigned)MT, 256, 0, stream>>>(tmp, gam + (long)l * DD,
                                                  bet + (long)l * DD, h, hb);

    // act = gelu(h @ W_in) -> bf16 [2048, 2048]
    k_gemm_lds<EP_GELU_BF><<<dim3(DMM / 64, MT / 128, 1), 128, 0, stream>>>(
        hb, DD, 0, 0, wi, DD, 0, 0, (int)MT, DMM, DD, 1,
        nullptr, act, DMM, 1, 0, 0, nullptr, 0.0f);

    // h = h + act @ W_out -> fp32 h + bf16 hb
    k_gemm_lds<EP_RES_F32_BF><<<dim3(DD / 64, MT / 128, 1), 128, 0, stream>>>(
        act, DMM, 0, 0, wo, DMM, 0, 0, (int)MT, DD, DMM, 1,
        h, hb, DD, 1, 0, 0, h, 0.0f);
  }

  // ---- unembed: out = h @ embed^T  (fp32 [B*S, V]) ----
  k_gemm_lds<EP_F32><<<dim3((VV + 63) / 64, MT / 128, 1), 128, 0, stream>>>(
      hb, DD, 0, 0, embB, DD, 0, 0, (int)MT, VV, DD, 1,
      out, nullptr, VV, 1, 0, 0, nullptr, 0.0f);
}
